// UnetUpBlock_18777597018409
// MI455X (gfx1250) — matrix operand order repaired and checked by reading, not compiled, not run on hardware
//
#include <hip/hip_runtime.h>
#include <hip/hip_bf16.h>
#include <math.h>

// ---------------- CDNA5 (gfx1250) wave32 WMMA pipeline ----------------
typedef __attribute__((ext_vector_type(16))) _Float16 v16h;
typedef __attribute__((ext_vector_type(8)))  _Float16 v8h;
typedef __attribute__((ext_vector_type(8)))  float    v8f;
typedef __attribute__((ext_vector_type(4)))  int      v4i;

#define AS1 __attribute__((address_space(1)))
#define AS3 __attribute__((address_space(3)))

#if __has_builtin(__builtin_amdgcn_global_load_async_to_lds_b128) && \
    __has_builtin(__builtin_amdgcn_s_wait_asynccnt)
#define HAVE_ASYNC 1
#else
#define HAVE_ASYNC 0
#endif

#define BATCH  2
#define CQ     128      // C_OUT
#define LTOK   4096     // 64*64
#define TTOK   8192     // BATCH*LTOK
#define DIN    256      // D_INNER
#define DST    16       // D_STATE
#define DTRANK 8
#define KPACK  384      // 256 (inp chans) + 128 (skip chans)
#define NXP    48       // DT_RANK + 2*D_STATE = 40, padded to 48
#define EPSN   1e-5f

// ---------------- weight prep ----------------
__global__ void k_cvt(const float* __restrict__ s, _Float16* __restrict__ d, int n) {
  int i = blockIdx.x * blockDim.x + threadIdx.x;
  if (i < n) d[i] = (_Float16)s[i];
}

__global__ void k_cvt_pad_wx(const float* __restrict__ s, _Float16* __restrict__ d) {
  int i = blockIdx.x * blockDim.x + threadIdx.x;
  if (i >= NXP * DIN) return;
  int row = i >> 8;
  d[i] = (row < 40) ? (_Float16)s[i] : (_Float16)0.0f;
}

// Wbig[par][o][k]: k<256: sum_c w_up[k][c][ph][pw]*w_fuse[c][o]; k>=256: w_fuse[k-128][o]
__global__ void k_wbig(const float* __restrict__ w_up, const float* __restrict__ w_fuse,
                       _Float16* __restrict__ Wbig) {
  int i = blockIdx.x * blockDim.x + threadIdx.x;
  if (i >= 4 * CQ * KPACK) return;
  int k = i % KPACK;
  int t = i / KPACK;
  int o = t % CQ;
  int par = t / CQ;
  int ph = par >> 1, pw = par & 1;
  float v;
  if (k < 256) {
    v = 0.f;
    for (int c = 0; c < CQ; ++c)
      v += w_up[((k * CQ + c) * 2 + ph) * 2 + pw] * w_fuse[c * CQ + o];
  } else {
    v = w_fuse[(k - CQ) * CQ + o];
  }
  Wbig[i] = (_Float16)v;
}

__global__ void k_negA(const float* __restrict__ alog, float* __restrict__ an) {
  int i = blockIdx.x * blockDim.x + threadIdx.x;
  if (i < DIN * DST) an[i] = -__expf(alog[i]);
}

// ---------------- activation pack (parity-grouped token order q) ----------------
__global__ void k_pack(const float* __restrict__ inp, const float* __restrict__ skip,
                       _Float16* __restrict__ Apack) {
  int idx = blockIdx.x * blockDim.x + threadIdx.x;
  if (idx >= TTOK * KPACK) return;
  int i  = idx % KPACK;
  int q  = idx / KPACK;
  int wq = q & 31;
  int g  = q >> 5;
  int pw = g & 1;
  int hh = (g >> 1) & 63;
  int b  = g >> 7;
  int w  = wq * 2 + pw;
  float v;
  if (i < 256) v = inp[((b * 256 + i) << 10) + ((hh >> 1) << 5) + (w >> 1)];
  else         v = skip[((b * CQ + (i - 256)) << 12) + (hh << 6) + w];
  Apack[idx] = (_Float16)v;
}

// ---------------- fused up+concat+fuse GEMM: 16x64 per wave, B global ----------------
__global__ void k_gemm_fused4(const _Float16* __restrict__ Apack,
                              const _Float16* __restrict__ Wbig,
                              float* __restrict__ fused) {
  int wid = blockIdx.x * 8 + (threadIdx.x >> 5);   // 1024 waves total
  int mtile = wid >> 1;
  int n0 = (wid & 1) * 64;
  int q0 = mtile * 16;
  int g  = q0 >> 5;
  int pw = g & 1, hh = (g >> 1) & 63, b = g >> 7;
  int par = ((hh & 1) << 1) | pw;
  const int lane = threadIdx.x & 31;
  const int row = lane & 15, half = lane >> 4;
  const _Float16* ap    = Apack + (q0 + row) * KPACK;
  const _Float16* bbase = Wbig + (par * CQ + n0 + row) * KPACK;
  v8f zero = {0.f,0.f,0.f,0.f,0.f,0.f,0.f,0.f};
  v8f acc[4];
#pragma unroll
  for (int j = 0; j < 4; ++j) acc[j] = zero;
  for (int k0 = 0; k0 < KPACK; k0 += 32) {
    if (k0 + 128 < KPACK) __builtin_prefetch(ap + k0 + 128, 0, 3);
    v8h a0 = *(const v8h*)(ap + k0 + half * 8);
    v8h a1 = *(const v8h*)(ap + k0 + 16 + half * 8);
    v16h av;
#pragma unroll
    for (int i = 0; i < 8; ++i) { av[i] = a0[i]; av[i + 8] = a1[i]; }
#pragma unroll
    for (int j = 0; j < 4; ++j) {
      const _Float16* bp = bbase + j * 16 * KPACK + k0 + half * 8;
      v8h b0 = *(const v8h*)bp;
      v8h b1 = *(const v8h*)(bp + 16);
      v16h bv;
#pragma unroll
      for (int i = 0; i < 8; ++i) { bv[i] = b0[i]; bv[i + 8] = b1[i]; }
      acc[j] = __builtin_amdgcn_wmma_f32_16x16x32_f16(false, av, false, bv,
                                                      (short)0, acc[j], false, false);
    }
  }
  const int rb = half * 8;
#pragma unroll
  for (int j = 0; j < 4; ++j) {
    int n = n0 + j * 16 + row;
#pragma unroll
    for (int r = 0; r < 8; ++r) {
      int q  = q0 + rb + r;
      int wq = q & 31;
      int p  = b * LTOK + hh * 64 + (wq * 2 + pw);
      fused[p * CQ + n] = acc[j][r];
    }
  }
}

// ---------- generic group GEMM: B panel staged to LDS (async), 16x(NT*16)/wave ----------
// grid.x = M/128, grid.y = N/(NT*16); 256 threads (8 waves).
template<int K, int NT, int EPI>
__global__ void k_gemm_lds(const _Float16* __restrict__ A,
                           const _Float16* __restrict__ Bt,
                           float* __restrict__ o0, float* __restrict__ o1,
                           _Float16* __restrict__ oh,
                           const float* __restrict__ e0,
                           const float* __restrict__ e1) {
  __shared__ _Float16 bpanel[NT * 16 * K];
  const int n0 = blockIdx.y * (NT * 16);
  const char* bsrc = (const char*)(Bt + n0 * K);
  char* bdst = (char*)bpanel;
  const int bytes = NT * 16 * K * 2;
#if HAVE_ASYNC
  for (int t = threadIdx.x * 16; t < bytes; t += 256 * 16)
    __builtin_amdgcn_global_load_async_to_lds_b128(
        (AS1 v4i*)(bsrc + t), (AS3 v4i*)(bdst + t), 0, 0);
  __builtin_amdgcn_s_wait_asynccnt(0);
#else
  for (int t = threadIdx.x * 16; t < bytes; t += 256 * 16)
    *(v8h*)(bdst + t) = *(const v8h*)(bsrc + t);
#endif
  __syncthreads();

  const int lane = threadIdx.x & 31;
  const int row = lane & 15, half = lane >> 4;
  const int mtile = blockIdx.x * 8 + (threadIdx.x >> 5);
  const _Float16* ap = A + (mtile * 16 + row) * K;
  v8f zero = {0.f,0.f,0.f,0.f,0.f,0.f,0.f,0.f};
  v8f acc[NT];
#pragma unroll
  for (int j = 0; j < NT; ++j) acc[j] = zero;
  for (int k0 = 0; k0 < K; k0 += 32) {
    if (k0 + 128 < K) __builtin_prefetch(ap + k0 + 128, 0, 3);
    v8h a0 = *(const v8h*)(ap + k0 + half * 8);
    v8h a1 = *(const v8h*)(ap + k0 + 16 + half * 8);
    v16h av;
#pragma unroll
    for (int i = 0; i < 8; ++i) { av[i] = a0[i]; av[i + 8] = a1[i]; }
#pragma unroll
    for (int j = 0; j < NT; ++j) {
      const _Float16* bp = bpanel + (j * 16 + row) * K + k0 + half * 8;
      v8h b0 = *(const v8h*)bp;
      v8h b1 = *(const v8h*)(bp + 16);
      v16h bv;
#pragma unroll
      for (int i = 0; i < 8; ++i) { bv[i] = b0[i]; bv[i + 8] = b1[i]; }
      acc[j] = __builtin_amdgcn_wmma_f32_16x16x32_f16(false, av, false, bv,
                                                      (short)0, acc[j], false, false);
    }
  }
  const int mb = mtile * 16 + half * 8;
#pragma unroll
  for (int j = 0; j < NT; ++j) {
    int n = n0 + j * 16 + row;
#pragma unroll
    for (int r = 0; r < 8; ++r) {
      int m = mb + r;
      float v = acc[j][r];
      if (EPI == 0) {            // tokens | gate z
        if (n < CQ) o0[m * CQ + n] = v;
        else {
          float u  = v + e0[n - CQ];
          float si = u / (1.f + __expf(-u));
          o1[m * CQ + (n - CQ)] = 1.f / (1.f + __expf(-si));
        }
      } else if (EPI == 1) {     // xm | silu(zm)
        if (n < DIN) o0[m * DIN + n] = v;
        else         o1[m * DIN + (n - DIN)] = v / (1.f + __expf(-v));
      } else if (EPI == 2) {     // x_dbl
        o0[m * NXP + n] = v;
      } else if (EPI == 3) {     // gh = f16((y @ Wout) * z)
        oh[m * CQ + n] = (_Float16)(v * e0[m * CQ + n]);
      } else {                   // final_pre = x + tokens + (g @ gW_out)
        o0[m * CQ + n] = e0[m * CQ + n] + e1[m * CQ + n] + v;
      }
    }
  }
}

// ---------------- norms / elementwise ----------------
__global__ void k_stats(const float* __restrict__ src, float* __restrict__ mean,
                        float* __restrict__ rstd) {
  int bc = blockIdx.x;              // b*128 + c
  int b = bc >> 7, c = bc & 127;
  __shared__ float s1[256], s2[256];
  float a = 0.f, q = 0.f;
  for (int l = threadIdx.x; l < LTOK; l += 256) {
    float v = src[(b * LTOK + l) * CQ + c];
    a += v; q += v * v;
  }
  s1[threadIdx.x] = a; s2[threadIdx.x] = q;
  __syncthreads();
  for (int st = 128; st > 0; st >>= 1) {
    if (threadIdx.x < st) { s1[threadIdx.x] += s1[threadIdx.x + st];
                            s2[threadIdx.x] += s2[threadIdx.x + st]; }
    __syncthreads();
  }
  if (threadIdx.x == 0) {
    float mu  = s1[0] / (float)LTOK;
    float var = s2[0] / (float)LTOK - mu * mu;
    mean[bc] = mu;
    rstd[bc] = rsqrtf(var + EPSN);
  }
}

__global__ void k_apply_x(const float* __restrict__ fused, const float* __restrict__ mean,
                          const float* __restrict__ rstd,
                          float* __restrict__ xf, _Float16* __restrict__ xh) {
  int idx = blockIdx.x * blockDim.x + threadIdx.x;
  if (idx >= TTOK * CQ) return;
  int c = idx & 127, p = idx >> 7;
  int bc = ((p >> 12) << 7) + c;
  float v = (fused[idx] - mean[bc]) * rstd[bc];
  float x = (v > 0.f) ? v : 0.01f * v;
  xf[idx] = x;
  xh[idx] = (_Float16)x;
}

__global__ void k_ln(const float* __restrict__ tokens, const float* __restrict__ g,
                     const float* __restrict__ bb, _Float16* __restrict__ th) {
  int p = blockIdx.x;
  int c = threadIdx.x;               // 128 threads
  __shared__ float s1[128], s2[128];
  float v = tokens[p * CQ + c];
  s1[c] = v; s2[c] = v * v;
  __syncthreads();
  for (int st = 64; st > 0; st >>= 1) {
    if (c < st) { s1[c] += s1[c + st]; s2[c] += s2[c + st]; }
    __syncthreads();
  }
  float mu  = s1[0] / (float)CQ;
  float var = s2[0] / (float)CQ - mu * mu;
  float r   = rsqrtf(var + EPSN);
  th[p * CQ + c] = (_Float16)((v - mu) * r * g[c] + bb[c]);
}

__global__ void k_conv(const float* __restrict__ xm, const float* __restrict__ cw,
                       const float* __restrict__ cb,
                       float* __restrict__ xcf, _Float16* __restrict__ xch) {
  int idx = blockIdx.x * blockDim.x + threadIdx.x;
  if (idx >= TTOK * DIN) return;
  int d = idx & 255, p = idx >> 8;
  int b = p >> 12, l = p & 4095;
  float acc = cb[d];
#pragma unroll
  for (int j = 0; j < 4; ++j) {
    int ll = l - 3 + j;
    if (ll >= 0) acc += xm[((b << 12) + ll) * DIN + d] * cw[d * 4 + j];
  }
  float s = acc / (1.f + __expf(-acc));
  xcf[idx] = s;
  xch[idx] = (_Float16)s;
}

__global__ void k_dt(const float* __restrict__ xdbl, const float* __restrict__ Wdt,
                     const float* __restrict__ bdt, float* __restrict__ dtb) {
  int idx = blockIdx.x * blockDim.x + threadIdx.x;
  if (idx >= TTOK * DIN) return;
  int d = idx & 255, p = idx >> 8;
  float u = bdt[d];
#pragma unroll
  for (int r = 0; r < DTRANK; ++r) u += xdbl[p * NXP + r] * Wdt[d * DTRANK + r];
  dtb[idx] = (u > 20.f) ? u : log1pf(__expf(u));
}

// selective scan: thread = one (d, s) state; 16 s-lanes reduce via shfl_xor
__global__ void k_scan(const float* __restrict__ dtb, const float* __restrict__ xcf,
                       const float* __restrict__ xdbl, const float* __restrict__ An,
                       float* __restrict__ y) {
  int b    = blockIdx.x >> 4;
  int dblk = blockIdx.x & 15;
  int d = dblk * 16 + (threadIdx.x >> 4);
  int s = threadIdx.x & 15;
  float a_ds = An[d * DST + s];
  float h = 0.f;
  for (int l = 0; l < LTOK; ++l) {
    int p = b * LTOK + l;
    float dtv = dtb[p * DIN + d];
    float xcv = xcf[p * DIN + d];
    float Bv  = xdbl[p * NXP + DTRANK + s];
    float Cv  = xdbl[p * NXP + DTRANK + DST + s];
    h = h * __expf(dtv * a_ds) + dtv * xcv * Bv;
    float yc = h * Cv;
    yc += __shfl_xor(yc, 1, 32);
    yc += __shfl_xor(yc, 2, 32);
    yc += __shfl_xor(yc, 4, 32);
    yc += __shfl_xor(yc, 8, 32);
    if (s == 0) y[p * DIN + d] = yc;
  }
}

__global__ void k_ymul(const float* __restrict__ y, const float* __restrict__ xcf,
                       const float* __restrict__ mD, const float* __restrict__ szm,
                       _Float16* __restrict__ yh) {
  int idx = blockIdx.x * blockDim.x + threadIdx.x;
  if (idx >= TTOK * DIN) return;
  int d = idx & 255;
  float v = y[idx] + xcf[idx] * mD[d];
  yh[idx] = (_Float16)(v * szm[idx]);
}

__global__ void k_out(const float* __restrict__ finalp, const float* __restrict__ mean,
                      const float* __restrict__ rstd, float* __restrict__ out) {
  int idx = blockIdx.x * blockDim.x + threadIdx.x;
  if (idx >= TTOK * CQ) return;
  int c = idx & 127, p = idx >> 7;
  int b = p >> 12, hw = p & 4095;
  int bc = (b << 7) + c;
  out[((b * CQ) + c) * LTOK + hw] = (finalp[idx] - mean[bc]) * rstd[bc];
}

// ---------------- host ----------------
extern "C" void kernel_launch(void* const* d_in, const int* in_sizes, int n_in,
                              void* d_out, int out_size, void* d_ws, size_t ws_size,
                              hipStream_t stream) {
  const float* inp    = (const float*)d_in[0];
  const float* skip   = (const float*)d_in[1];
  const float* w_up   = (const float*)d_in[2];
  const float* w_fuse = (const float*)d_in[3];
  const float* gW_in  = (const float*)d_in[4];
  const float* gW_gate= (const float*)d_in[5];
  const float* gb_gate= (const float*)d_in[6];
  const float* ln_g   = (const float*)d_in[7];
  const float* ln_b   = (const float*)d_in[8];
  const float* gW_out = (const float*)d_in[9];
  const float* m_Win  = (const float*)d_in[10];
  const float* m_convw= (const float*)d_in[11];
  const float* m_convb= (const float*)d_in[12];
  const float* m_Wx   = (const float*)d_in[13];
  const float* m_Wdt  = (const float*)d_in[14];
  const float* m_bdt  = (const float*)d_in[15];
  const float* m_Alog = (const float*)d_in[16];
  const float* m_D    = (const float*)d_in[17];
  const float* m_Wout = (const float*)d_in[18];
  float* out = (float*)d_out;

  char* ws = (char*)d_ws;
  size_t off = 0;
  auto alloc = [&](size_t bytes) -> void* {
    void* p = ws + off;
    off += (bytes + 255) & ~(size_t)255;
    return p;
  };

  _Float16* Wbig   = (_Float16*)alloc(4 * CQ * KPACK * sizeof(_Float16));
  _Float16* gWcat  = (_Float16*)alloc(256 * CQ * sizeof(_Float16));
  _Float16* Win_h  = (_Float16*)alloc(512 * CQ * sizeof(_Float16));
  _Float16* Wx_h   = (_Float16*)alloc(NXP * DIN * sizeof(_Float16));
  _Float16* Wout_h = (_Float16*)alloc(CQ * DIN * sizeof(_Float16));
  _Float16* gWo_h  = (_Float16*)alloc(CQ * CQ * sizeof(_Float16));
  float*    An     = (float*)alloc(DIN * DST * sizeof(float));
  _Float16* Apack  = (_Float16*)alloc((size_t)TTOK * KPACK * sizeof(_Float16));
  float*    fused  = (float*)alloc((size_t)TTOK * CQ * sizeof(float));
  float*    xf     = (float*)alloc((size_t)TTOK * CQ * sizeof(float));
  _Float16* xh     = (_Float16*)alloc((size_t)TTOK * CQ * sizeof(_Float16));
  float*    tokens = (float*)alloc((size_t)TTOK * CQ * sizeof(float));
  float*    zbuf   = (float*)alloc((size_t)TTOK * CQ * sizeof(float));
  _Float16* th     = (_Float16*)alloc((size_t)TTOK * CQ * sizeof(_Float16));
  float*    xm     = (float*)alloc((size_t)TTOK * DIN * sizeof(float));
  float*    szm    = (float*)alloc((size_t)TTOK * DIN * sizeof(float));
  float*    xcf    = (float*)alloc((size_t)TTOK * DIN * sizeof(float));
  _Float16* xch    = (_Float16*)alloc((size_t)TTOK * DIN * sizeof(_Float16));
  float*    xdbl   = (float*)alloc((size_t)TTOK * NXP * sizeof(float));
  float*    dtb    = (float*)alloc((size_t)TTOK * DIN * sizeof(float));
  float*    ybuf   = (float*)alloc((size_t)TTOK * DIN * sizeof(float));
  _Float16* yh     = (_Float16*)alloc((size_t)TTOK * DIN * sizeof(_Float16));
  _Float16* gh     = (_Float16*)alloc((size_t)TTOK * CQ * sizeof(_Float16));
  float*    mean1  = (float*)alloc(256 * sizeof(float));
  float*    rstd1  = (float*)alloc(256 * sizeof(float));
  float*    mean2  = (float*)alloc(256 * sizeof(float));
  float*    rstd2  = (float*)alloc(256 * sizeof(float));
  float*    finalp = fused;   // reuse: fused dead after k_apply_x

  const int TB = 256;
  // weights
  k_cvt<<<(CQ*CQ + TB-1)/TB, TB, 0, stream>>>(gW_in,  gWcat,            CQ*CQ);
  k_cvt<<<(CQ*CQ + TB-1)/TB, TB, 0, stream>>>(gW_gate,gWcat + CQ*CQ,    CQ*CQ);
  k_cvt<<<(512*CQ + TB-1)/TB, TB, 0, stream>>>(m_Win, Win_h,            512*CQ);
  k_cvt<<<(CQ*DIN + TB-1)/TB, TB, 0, stream>>>(m_Wout,Wout_h,           CQ*DIN);
  k_cvt<<<(CQ*CQ + TB-1)/TB, TB, 0, stream>>>(gW_out, gWo_h,            CQ*CQ);
  k_cvt_pad_wx<<<(NXP*DIN + TB-1)/TB, TB, 0, stream>>>(m_Wx, Wx_h);
  k_wbig<<<(4*CQ*KPACK + TB-1)/TB, TB, 0, stream>>>(w_up, w_fuse, Wbig);
  k_negA<<<(DIN*DST + TB-1)/TB, TB, 0, stream>>>(m_Alog, An);

  // stage 1: upsample+concat+fuse as one WMMA GEMM, then instance norm + lrelu
  k_pack<<<(TTOK*KPACK)/TB, TB, 0, stream>>>(inp, skip, Apack);
  k_gemm_fused4<<<128, TB, 0, stream>>>(Apack, Wbig, fused);
  k_stats<<<256, TB, 0, stream>>>(fused, mean1, rstd1);
  k_apply_x<<<(TTOK*CQ)/TB, TB, 0, stream>>>(fused, mean1, rstd1, xf, xh);

  // stage 2: mixer front GEMMs (B panels staged to LDS via async)
  k_gemm_lds<128,4,0><<<dim3(64,4), TB, 0, stream>>>(xh, gWcat, tokens, zbuf,
                                                     nullptr, gb_gate, nullptr);
  k_ln<<<TTOK, 128, 0, stream>>>(tokens, ln_g, ln_b, th);
  k_gemm_lds<128,4,1><<<dim3(64,8), TB, 0, stream>>>(th, Win_h, xm, szm,
                                                     nullptr, nullptr, nullptr);
  k_conv<<<(TTOK*DIN)/TB, TB, 0, stream>>>(xm, m_convw, m_convb, xcf, xch);
  k_gemm_lds<256,3,2><<<dim3(64,1), TB, 0, stream>>>(xch, Wx_h, xdbl, nullptr,
                                                     nullptr, nullptr, nullptr);
  k_dt<<<(TTOK*DIN)/TB, TB, 0, stream>>>(xdbl, m_Wdt, m_bdt, dtb);

  // stage 3: selective scan (sequential over L)
  k_scan<<<BATCH * (DIN/16), TB, 0, stream>>>(dtb, xcf, xdbl, An, ybuf);
  k_ymul<<<(TTOK*DIN)/TB, TB, 0, stream>>>(ybuf, xcf, m_D, szm, yh);

  // stage 4: output GEMMs + residual + final instance norm
  k_gemm_lds<256,4,3><<<dim3(64,2), TB, 0, stream>>>(yh, Wout_h, nullptr, nullptr,
                                                     gh, zbuf, nullptr);
  k_gemm_lds<128,4,4><<<dim3(64,2), TB, 0, stream>>>(gh, gWo_h, finalp, nullptr,
                                                     nullptr, xf, tokens);
  k_stats<<<256, TB, 0, stream>>>(finalp, mean2, rstd2);
  k_out<<<(TTOK*CQ)/TB, TB, 0, stream>>>(finalp, mean2, rstd2, out);
  (void)in_sizes; (void)n_in; (void)out_size; (void)ws_size;
}
